// EinsumTransformer_12317966205220
// MI455X (gfx1250) — matrix operand order repaired and checked by reading, not compiled, not run on hardware
//
#include <hip/hip_runtime.h>
#include <math.h>

// CDNA5 (gfx1250) tiny-transformer batch kernel, v2.
// - 2 batch elements per wave-iteration -> 16-row M tile; K=32 == D matches
//   V_WMMA_F32_16X16X32_F16 exactly. All GEMMs via WMMA.
// - All 49 weight B-tiles staged ONCE per block into LDS, f16, pre-swizzled
//   into WMMA B-fragment lane layout -> hot loop B loads are 2x ds_load_b128.
// - Each wave loops over IT=4 batch-pairs to amortize staging.

typedef __attribute__((ext_vector_type(16))) _Float16 v16h;
typedef __attribute__((ext_vector_type(8)))  float    v8f;

#define DMODEL 32
#define SEQ    6
#define NW     4      // waves per block (128 threads, wave32)
#define BPW    2      // batch elements per wave-iteration
#define IT     4      // batch-pair iterations per wave
#define ROWS   16     // 2 batch * 8 padded seq
#define DFF    128
#define NTILES 49     // 2 layers * 24 weight tiles + 1 logits tile
#define FRAGSZ 512    // 32 lanes * 16 halves per B-fragment tile

// ---- WMMA fragment helpers (CDNA5 ISA 7.12.2 layouts) ----

// A matrix 16x32 f16: lane L -> M = L&15, half hs = L>>4.
// element e -> K = (e>>3)*16 + hs*8 + ((e>>1)&3)*2 + (e&1)
// (per lane this is two contiguous 8-half runs -> 2x ds_load_b128)
__device__ __forceinline__ v16h load_A16(const _Float16* t, int lane, int ld) {
  const int M = lane & 15, hs = lane >> 4;
  v16h a;
#pragma unroll
  for (int e = 0; e < 16; ++e) {
    int K = ((e >> 3) << 4) + hs * 8 + (((e >> 1) & 3) << 1) + (e & 1);
    a[e] = t[M * ld + K];
  }
  return a;
}

// Pre-swizzled B fragment: 32 contiguous bytes per lane.
__device__ __forceinline__ v16h load_Bf(const _Float16* Wfrag, int t, int lane) {
  return *(const v16h*)(Wfrag + t * FRAGSZ + lane * 16);
}

// C/D 16x16 f32: lane L -> N = L&15; VGPR r -> M = r + 8*(L>>4)
__device__ __forceinline__ v8f load_C(const float* t, int lane, int ld, int ncol) {
  const int N = (lane & 15) + ncol, hs = lane >> 4;
  v8f c;
#pragma unroll
  for (int r = 0; r < 8; ++r) c[r] = t[(r + 8 * hs) * ld + N];
  return c;
}

__device__ __forceinline__ void store_D(float* t, v8f d, int lane, int ld, int ncol) {
  const int N = (lane & 15) + ncol, hs = lane >> 4;
#pragma unroll
  for (int r = 0; r < 8; ++r) t[(r + 8 * hs) * ld + N] = d[r];
}

__device__ __forceinline__ void store_Dh(_Float16* t, v8f d, int lane, int ld, int ncol) {
  const int N = (lane & 15) + ncol, hs = lane >> 4;
#pragma unroll
  for (int r = 0; r < 8; ++r) t[(r + 8 * hs) * ld + N] = (_Float16)d[r];
}

__device__ __forceinline__ v8f wmma_f16(v16h a, v16h b, v8f c) {
  return __builtin_amdgcn_wmma_f32_16x16x32_f16(false, a, false, b, (short)0, c,
                                                false, false);
}

// LayerNorm of a 16x32 f32 tile -> f16 tile. Lanes 0..15 each own one row.
// Pad rows are all-zero -> LN(0)=0, stays zero.
__device__ __forceinline__ void layernorm_tile(const float* X, _Float16* Xn, int lane) {
  if (lane < ROWS) {
    const float* xr = X + lane * DMODEL;
    float mu = 0.f;
#pragma unroll
    for (int d = 0; d < DMODEL; ++d) mu += xr[d];
    mu *= (1.f / DMODEL);
    float var = 0.f;
#pragma unroll
    for (int d = 0; d < DMODEL; ++d) { float t = xr[d] - mu; var += t * t; }
    var *= (1.f / DMODEL);
    float r = rsqrtf(var + 1e-5f);
    _Float16* xo = Xn + lane * DMODEL;
#pragma unroll
    for (int d = 0; d < DMODEL; ++d) xo[d] = (_Float16)((xr[d] - mu) * r);
  }
}

__global__ __launch_bounds__(NW * 32) void tinyformer_kernel(
    const int* __restrict__ tokens, const float* __restrict__ tok_emb,
    const float* __restrict__ pos_emb, const float* __restrict__ Wq,
    const float* __restrict__ Wk, const float* __restrict__ Wv,
    const float* __restrict__ Wo, const float* __restrict__ W1,
    const float* __restrict__ W2, float* __restrict__ out, int Btot) {
  // Weight B-fragments, f16, pre-swizzled. Tile numbering per layer L (base L*24):
  //  +0..1 Wq(nt) | +2..3 Wk | +4..5 Wv | +6..7 Wo | +8..15 W1(nt) |
  //  +16..23 W2(kc*2+nt) | tile 48 = logits (tok_emb^T, N<15 valid)
  __shared__ __align__(32) _Float16 Wfrag[NTILES * FRAGSZ];

  __shared__ float    Xs[NW][ROWS * DMODEL];   // residual stream (f32)
  __shared__ _Float16 Xnh[NW][ROWS * DMODEL];  // LN output (f16 A tile)
  __shared__ float    Qs[NW][ROWS * DMODEL];
  __shared__ float    Ks_[NW][ROWS * DMODEL];
  __shared__ float    Vs[NW][ROWS * DMODEL];
  __shared__ _Float16 Ohs[NW][ROWS * DMODEL];  // attention out (f16 A tile)
  __shared__ _Float16 Hh[NW][ROWS * DFF];      // MLP hidden (f16 A tile)

  const int lane = threadIdx.x & 31;
  const int wv   = threadIdx.x >> 5;

  // ---- one-time: stage + swizzle + convert all weight tiles ----
  for (int idx = threadIdx.x; idx < NTILES * FRAGSZ; idx += NW * 32) {
    const int t  = idx >> 9;          // tile id
    const int rr = idx & 511;
    const int fl = rr >> 4;           // fragment lane
    const int e  = rr & 15;           // element within lane
    const int N  = fl & 15, ks = fl >> 4;
    const int K  = (e >> 3) * 16 + ks * 8 + (e & 7);
    _Float16 v;
    if (t == 48) {
      v = (N < 15) ? (_Float16)tok_emb[N * DMODEL + K] : (_Float16)0.f;
    } else {
      const int L = t / 24, r = t % 24;
      const float* src;
      int ld, nc;
      if (r < 8) {
        const int w = r >> 1;
        src = (w == 0 ? Wq : w == 1 ? Wk : w == 2 ? Wv : Wo) + L * DMODEL * DMODEL;
        ld  = DMODEL;
        nc  = (r & 1) * 16;
      } else if (r < 16) {
        src = W1 + L * DMODEL * DFF;
        ld  = DFF;
        nc  = (r - 8) * 16;
      } else {
        const int j = r - 16, kc = j >> 1, nt = j & 1;
        src = W2 + L * DFF * DMODEL + kc * DMODEL * DMODEL;
        ld  = DMODEL;
        nc  = nt * 16;
      }
      v = (_Float16)src[K * ld + nc + N];
    }
    Wfrag[t * FRAGSZ + fl * 16 + e] = v;
  }
  __syncthreads();

  for (int it = 0; it < IT; ++it) {
    const long base = ((long)blockIdx.x * NW + wv) * (BPW * IT) + (long)it * BPW;

    // ---- embedding: X[m][c] = tok_emb[tok] + pos_emb[s]; pad rows = 0 ----
    for (int m = 0; m < ROWS; ++m) {
      const int  rb = m >> 3, s = m & 7;
      const long b  = base + rb;
      float      v  = 0.f;
      if (s < SEQ && b < (long)Btot) {
        const int tok = tokens[b * SEQ + s];
        v = tok_emb[tok * DMODEL + lane] + pos_emb[s * DMODEL + lane];
      }
      Xs[wv][m * DMODEL + lane] = v;
    }
    __syncthreads();

    for (int L = 0; L < 2; ++L) {
      const int Lb = L * 24;

      // ---- LN1 ----
      layernorm_tile(Xs[wv], Xnh[wv], lane);
      __syncthreads();

      // ---- Q,K,V projections: 3 x (2 WMMA) ----
      {
        v16h a = load_A16(Xnh[wv], lane, DMODEL);
#pragma unroll
        for (int nt = 0; nt < 2; ++nt) {
          v8f c = {};
          c = wmma_f16(a, load_Bf(Wfrag, Lb + 0 + nt, lane), c);
          store_D(Qs[wv], c, lane, DMODEL, nt * 16);
          v8f ck = {};
          ck = wmma_f16(a, load_Bf(Wfrag, Lb + 2 + nt, lane), ck);
          store_D(Ks_[wv], ck, lane, DMODEL, nt * 16);
          v8f cv = {};
          cv = wmma_f16(a, load_Bf(Wfrag, Lb + 4 + nt, lane), cv);
          store_D(Vs[wv], cv, lane, DMODEL, nt * 16);
        }
      }
      __syncthreads();

      // ---- zero O tile (covers pad rows), then attention in VALU ----
      for (int i = lane; i < ROWS * DMODEL; i += 32) Ohs[wv][i] = (_Float16)0.f;
      __syncthreads();

      // one lane per (batch, head, query) triple: 2*4*6 = 48 triples
      for (int i = lane; i < BPW * 4 * SEQ; i += 32) {
        const int rb = i / (4 * SEQ);
        const int rem = i % (4 * SEQ);
        const int h = rem / SEQ, q = rem % SEQ;
        const int row = rb * 8 + q;
        const float scale = 0.35355339059327373f;  // 1/sqrt(8)
        float qv[8];
#pragma unroll
        for (int c = 0; c < 8; ++c) qv[c] = Qs[wv][row * DMODEL + h * 8 + c];
        float sc[SEQ];
        float mx = -3.4e38f;
        for (int t = 0; t <= q; ++t) {  // causal: keys t <= q
          float s = 0.f;
#pragma unroll
          for (int c = 0; c < 8; ++c)
            s += qv[c] * Ks_[wv][(rb * 8 + t) * DMODEL + h * 8 + c];
          s *= scale;
          sc[t] = s;
          mx = fmaxf(mx, s);
        }
        float sum = 0.f;
        for (int t = 0; t <= q; ++t) { sc[t] = __expf(sc[t] - mx); sum += sc[t]; }
        const float inv = 1.f / sum;
        float o[8] = {0.f, 0.f, 0.f, 0.f, 0.f, 0.f, 0.f, 0.f};
        for (int t = 0; t <= q; ++t) {
#pragma unroll
          for (int c = 0; c < 8; ++c)
            o[c] += sc[t] * Vs[wv][(rb * 8 + t) * DMODEL + h * 8 + c];
        }
#pragma unroll
        for (int c = 0; c < 8; ++c)
          Ohs[wv][row * DMODEL + h * 8 + c] = (_Float16)(o[c] * inv);
      }
      __syncthreads();

      // ---- X += O @ Wo (residual folded into accumulator) ----
      {
        v16h ao = load_A16(Ohs[wv], lane, DMODEL);
#pragma unroll
        for (int nt = 0; nt < 2; ++nt) {
          v8f c = load_C(Xs[wv], lane, DMODEL, nt * 16);
          c = wmma_f16(ao, load_Bf(Wfrag, Lb + 6 + nt, lane), c);
          store_D(Xs[wv], c, lane, DMODEL, nt * 16);
        }
      }
      __syncthreads();

      // ---- LN2 ----
      layernorm_tile(Xs[wv], Xnh[wv], lane);
      __syncthreads();

      // ---- HID = gelu(Xn @ W1): 8 N-tiles ----
      {
        v16h a = load_A16(Xnh[wv], lane, DMODEL);
#pragma unroll
        for (int nt = 0; nt < 8; ++nt) {
          v8f c = {};
          c = wmma_f16(a, load_Bf(Wfrag, Lb + 8 + nt, lane), c);
#pragma unroll
          for (int r = 0; r < 8; ++r) {
            float x = c[r];
            c[r] = 0.5f * x * (1.0f + erff(x * 0.70710678118654752f));  // exact gelu
          }
          store_Dh(Hh[wv], c, lane, DFF, nt * 16);
        }
      }
      __syncthreads();

      // ---- X += HID @ W2: 2 N-tiles x 4 K-chunks of 32 ----
#pragma unroll
      for (int nt = 0; nt < 2; ++nt) {
        v8f c = load_C(Xs[wv], lane, DMODEL, nt * 16);
#pragma unroll
        for (int kc = 0; kc < 4; ++kc) {
          v16h a2 = load_A16(Hh[wv] + kc * DMODEL, lane, DFF);
          c = wmma_f16(a2, load_Bf(Wfrag, Lb + 16 + kc * 2 + nt, lane), c);
        }
        store_D(Xs[wv], c, lane, DMODEL, nt * 16);
      }
      __syncthreads();
    }

    // ---- final LN ----
    layernorm_tile(Xs[wv], Xnh[wv], lane);
    __syncthreads();

    // ---- logits = Xn @ tok_emb^T : [16x32] x [32x15(pad16)] -> 1 WMMA ----
    {
      v16h a = load_A16(Xnh[wv], lane, DMODEL);
      v8f  c = {};
      c = wmma_f16(a, load_Bf(Wfrag, 48, lane), c);
      const int N = lane & 15, hs = lane >> 4;
#pragma unroll
      for (int r = 0; r < 8; ++r) {
        const int  m = r + 8 * hs;
        const int  rb = m >> 3, s = m & 7;
        const long b = base + rb;
        if (s < SEQ && N < 15 && b < (long)Btot)
          out[(b * SEQ + s) * 15 + N] = c[r];
      }
    }
    __syncthreads();
  }
}

extern "C" void kernel_launch(void* const* d_in, const int* in_sizes, int n_in,
                              void* d_out, int out_size, void* d_ws, size_t ws_size,
                              hipStream_t stream) {
  const int*   tokens  = (const int*)d_in[0];
  const float* tok_emb = (const float*)d_in[1];
  const float* pos_emb = (const float*)d_in[2];
  const float* Wq      = (const float*)d_in[3];
  const float* Wk      = (const float*)d_in[4];
  const float* Wv      = (const float*)d_in[5];
  const float* Wo      = (const float*)d_in[6];
  const float* W1      = (const float*)d_in[7];
  const float* W2      = (const float*)d_in[8];
  float*       out     = (float*)d_out;

  const int Btot   = in_sizes[0] / SEQ;        // tokens is [B, SEQ]
  const int perBlk = NW * BPW * IT;            // 32 batch elements / block
  const int blocks = (Btot + perBlk - 1) / perBlk;

  hipLaunchKernelGGL(tinyformer_kernel, dim3(blocks), dim3(NW * 32), 0, stream,
                     tokens, tok_emb, pos_emb, Wq, Wk, Wv, Wo, W1, W2, out, Btot);
}
